// AttnBFAN_89361089561148
// MI455X (gfx1250) — matrix operand order repaired and checked by reading, not compiled, not run on hardware
//
#include <hip/hip_runtime.h>
#include <hip/hip_bf16.h>

// Problem constants (from reference setup_inputs)
#define B_ 128
#define Q_ 128
#define C_ 1024
#define D_ 1024
#define SMOOTH_ 20.0f
#define EPS_ 1e-8f

typedef __attribute__((ext_vector_type(16))) __bf16 bf16x16;
typedef __attribute__((ext_vector_type(8)))  float  v8f;
typedef __attribute__((ext_vector_type(8)))  int    v8i;

static __device__ __forceinline__ bf16x16 asbf(v8i v) {
    return __builtin_bit_cast(bf16x16, v);
}

// Pack bf16(a) (low 16) | bf16(b) (high 16) with round-to-nearest:
// two v_add_nc_u32 + one v_perm_b32.
static __device__ __forceinline__ int pk2(float a, float b) {
    unsigned ua = __builtin_bit_cast(unsigned, a) + 0x8000u;
    unsigned ub = __builtin_bit_cast(unsigned, b) + 0x8000u;
    return (int)__builtin_amdgcn_perm(ub, ua, 0x07060302u);
}

// scalar f32 -> bf16 bits, round-to-nearest
static __device__ __forceinline__ unsigned short bfu(float f) {
    return (unsigned short)((__builtin_bit_cast(unsigned, f) + 0x8000u) >> 16);
}

// Async copy of one 16B chunk per lane: global -> LDS (ASYNCcnt-tracked).
static __device__ __forceinline__ void async_ld16(unsigned lds_byte_off, const float* gptr) {
    asm volatile("global_load_async_to_lds_b128 %0, %1, off"
                 :: "v"(lds_byte_off), "v"((unsigned long long)(uintptr_t)gptr)
                 : "memory");
}
static __device__ __forceinline__ void wait_async0() {
    asm volatile("s_wait_asynccnt 0" ::: "memory");
}
static __device__ __forceinline__ unsigned lds_off(const void* p) {
    return (unsigned)(uintptr_t)p;  // low 32 bits of generic addr == LDS offset
}

// A-matrix 16x32 bf16 fragment, split into raw global loads (issue early,
// before barriers) and packing (consume late). Caller passes p = row + k0 + hi*16.
struct AFrag { float4 x0, x1, x2, x3; };
static __device__ __forceinline__ AFrag load_fragA_raw(const float* __restrict__ p) {
    AFrag f;
    f.x0 = *(const float4*)(p);
    f.x1 = *(const float4*)(p + 4);
    f.x2 = *(const float4*)(p + 8);
    f.x3 = *(const float4*)(p + 12);
    return f;
}
static __device__ __forceinline__ bf16x16 pack_fragA(const AFrag& f) {
    v8i r;
    r[0] = pk2(f.x0.x, f.x0.y); r[1] = pk2(f.x0.z, f.x0.w);
    r[2] = pk2(f.x1.x, f.x1.y); r[3] = pk2(f.x1.z, f.x1.w);
    r[4] = pk2(f.x2.x, f.x2.y); r[5] = pk2(f.x2.z, f.x2.w);
    r[6] = pk2(f.x3.x, f.x3.y); r[7] = pk2(f.x3.z, f.x3.w);
    return asbf(r);
}

// B-matrix 32x16 fragment from pre-packed bf16 in LDS: lane l (n=l&15, hi=l>>4)
// holds K = hi*16 + {0..15} contiguous -> two aligned ds_load_b128.
static __device__ __forceinline__ bf16x16 load_fragB_lds(const unsigned short* p) {
    const uint4 lo = *(const uint4*)(p);
    const uint4 hi = *(const uint4*)(p + 8);
    v8i r;
    r[0] = (int)lo.x; r[1] = (int)lo.y; r[2] = (int)lo.z; r[3] = (int)lo.w;
    r[4] = (int)hi.x; r[5] = (int)hi.y; r[6] = (int)hi.z; r[7] = (int)hi.w;
    return asbf(r);
}

// ---------------------------------------------------------------------------
// Kernel 1: attn[b,c,q] = leaky_relu(context[b,c,:] . query[b,q,:])
// Stored transposed as attnT (B,Q,C). Also norm2[b,c] = sum_q attn^2.
// Grid (C/128, B), block 256 (8 waves). Wave w owns c-strip [cb,cb+16) x all Q.
// Query tile (128q x 32k) async-staged to LDS (software-pipelined one tile
// ahead: raw chunks are per-thread private, so next-tile issue needs no sync).
// ---------------------------------------------------------------------------
__global__ __launch_bounds__(256) void gemm1_attn(
    const float* __restrict__ query, const float* __restrict__ context,
    float* __restrict__ attnT, float* __restrict__ norm2) {
    __shared__ __align__(16) float          rawQ[128 * 32];   // [q][k] f32 (16KB)
    __shared__ __align__(16) unsigned short ldsQ[128 * 32];   // [q][k] bf16 (8KB)

    const int t    = threadIdx.x;
    const int lane = t & 31;
    const int w    = t >> 5;
    const int b    = blockIdx.y;
    const int cb   = blockIdx.x * 128 + w * 16;
    const int lm   = lane & 15;
    const int hi   = lane >> 4;

    const float* arow  = context + ((size_t)b * C_ + cb + lm) * D_;
    const float* qbase = query + (size_t)b * Q_ * D_;
    const unsigned rawQ_base = lds_off(rawQ);

    auto stage_issue = [&](int kbase) {
#pragma unroll
        for (int it = 0; it < 4; ++it) {
            const int e  = it * 256 + t;        // 16B chunk id, 0..1023
            const int q  = e >> 3;              // 8 chunks per q row
            const int kc = (e & 7) * 4;
            async_ld16(rawQ_base + (unsigned)e * 16u,
                       qbase + (size_t)q * D_ + kbase + kc);
        }
    };

    v8f acc[8];
#pragma unroll
    for (int i = 0; i < 8; ++i) acc[i] = (v8f)(0.0f);

    stage_issue(0);                             // pipeline prologue
    for (int k0 = 0; k0 < D_; k0 += 32) {
        const AFrag araw = load_fragA_raw(arow + k0 + hi * 16);  // hoisted global
        wait_async0();                          // own raw chunks landed
        __syncthreads();                        // everyone done reading old ldsQ
        // pack own chunks f32 -> bf16 into ldsQ (values land in VGPRs here)
#pragma unroll
        for (int it = 0; it < 4; ++it) {
            const int e  = it * 256 + t;
            const int q  = e >> 3;
            const int kc = (e & 7) * 4;
            const float4 v = *(const float4*)(rawQ + e * 4);
            unsigned* wp = (unsigned*)(ldsQ + q * 32 + kc);
            wp[0] = (unsigned)pk2(v.x, v.y);
            wp[1] = (unsigned)pk2(v.z, v.w);
        }
        // raw chunks consumed into registers -> safe to refill for next tile
        if (k0 + 32 < D_) stage_issue(k0 + 32); // overlaps with compute below
        __syncthreads();

        const bf16x16 a = pack_fragA(araw);
        bf16x16 bb = load_fragB_lds(ldsQ + lm * 32 + hi * 16);
#pragma unroll
        for (int qt = 0; qt < 8; ++qt) {        // rotate B frag: load qt+1 early
            bf16x16 nxt = bb;
            if (qt < 7)
                nxt = load_fragB_lds(ldsQ + ((qt + 1) * 16 + lm) * 32 + hi * 16);
            acc[qt] = __builtin_amdgcn_wmma_f32_16x16x32_bf16(
                false, a, false, bb, (short)0, acc[qt], false, false);
            bb = nxt;
        }
    }

    // leaky relu, store attnT (q-major), accumulate per-c sum of squares
    float s[8];
#pragma unroll
    for (int r = 0; r < 8; ++r) s[r] = 0.0f;
#pragma unroll
    for (int qt = 0; qt < 8; ++qt) {
        float vals[8];
#pragma unroll
        for (int r = 0; r < 8; ++r) {
            float v = acc[qt][r];               // element (m = r+8*hi, n = lm)
            v = v > 0.0f ? v : 0.1f * v;
            vals[r] = v;
            s[r] += v * v;
        }
        const int q = qt * 16 + lm;
        float* dst = attnT + ((size_t)b * Q_ + q) * C_ + cb + hi * 8;
        *(float4*)(dst)     = make_float4(vals[0], vals[1], vals[2], vals[3]);
        *(float4*)(dst + 4) = make_float4(vals[4], vals[5], vals[6], vals[7]);
    }
    // reduce over the 16 lanes of each half (same hi -> same set of c rows)
#pragma unroll
    for (int off = 1; off < 16; off <<= 1) {
#pragma unroll
        for (int r = 0; r < 8; ++r) s[r] += __shfl_xor(s[r], off, 32);
    }
    if (lm == 0) {
#pragma unroll
        for (int r = 0; r < 8; ++r)
            norm2[(size_t)b * C_ + cb + hi * 8 + r] = s[r];
    }
}

// ---------------------------------------------------------------------------
// Kernel 2: per (b,q) row over C: x = attn / (sqrt(norm2)+eps) * 20,
// softmax, focal mask (> row mean), renormalize -> re_attn (B,Q,C).
// ---------------------------------------------------------------------------
__global__ __launch_bounds__(256) void softmax_focal(
    const float* __restrict__ attnT, const float* __restrict__ norm2,
    float* __restrict__ re_attn) {
    __shared__ float red[256];
    const int row = blockIdx.x;        // b*Q + q
    const int b   = row >> 7;          // Q_ == 128
    const int t   = threadIdx.x;
    const float* src = attnT + (size_t)row * C_;
    const float* n2  = norm2 + (size_t)b * C_;

    float x[4];
    float mx = -3.4e38f;
#pragma unroll
    for (int i = 0; i < 4; ++i) {
        const int c = t + i * 256;
        const float nv = sqrtf(n2[c]) + EPS_;
        const float v  = (src[c] / nv) * SMOOTH_;
        x[i] = v;
        mx = fmaxf(mx, v);
    }
    red[t] = mx; __syncthreads();
    for (int off = 128; off > 0; off >>= 1) {
        if (t < off) red[t] = fmaxf(red[t], red[t + off]);
        __syncthreads();
    }
    mx = red[0]; __syncthreads();

    float sum = 0.0f;
#pragma unroll
    for (int i = 0; i < 4; ++i) { x[i] = __expf(x[i] - mx); sum += x[i]; }
    red[t] = sum; __syncthreads();
    for (int off = 128; off > 0; off >>= 1) {
        if (t < off) red[t] += red[t + off];
        __syncthreads();
    }
    const float inv = 1.0f / red[0]; __syncthreads();

    float tsum = 0.0f;
#pragma unroll
    for (int i = 0; i < 4; ++i) {
        const float a = x[i] * inv;                       // softmax value
        x[i] = (a * (float)C_ - 1.0f > 0.0f) ? a : 0.0f;  // focal_equal mask
        tsum += x[i];
    }
    red[t] = tsum; __syncthreads();
    for (int off = 128; off > 0; off >>= 1) {
        if (t < off) red[t] += red[t + off];
        __syncthreads();
    }
    const float inv2 = 1.0f / red[0];
    float* dst = re_attn + (size_t)row * C_;
#pragma unroll
    for (int i = 0; i < 4; ++i) dst[t + i * 256] = x[i] * inv2;
}

// ---------------------------------------------------------------------------
// Kernel 3: wcontext[b,q,d] = sum_c re_attn[b,q,c] * context[b,c,d]
// M=Q=128 (8 waves x 16 rows), N tile = 128 d-cols, K loop over C=1024.
// Context tile async-staged (one tile ahead), transposed/packed to bf16 [n][k]
// (80B row stride, 16B aligned) so B-frags are two ds_load_b128.
// ---------------------------------------------------------------------------
__global__ __launch_bounds__(256) void gemm2_wctx(
    const float* __restrict__ re_attn, const float* __restrict__ context,
    float* __restrict__ wctx) {
    __shared__ __align__(16) float          rawC[32 * 128];   // [k][d] f32 (16KB)
    __shared__ __align__(16) unsigned short ldsBT[128 * 40];  // [n][k] bf16 (10KB)

    const int t    = threadIdx.x;
    const int lane = t & 31;
    const int w    = t >> 5;
    const int b    = blockIdx.y;
    const int db   = blockIdx.x * 128;
    const int lm   = lane & 15;
    const int hi   = lane >> 4;

    const float* arow = re_attn + ((size_t)b * Q_ + w * 16 + lm) * C_;
    const float* ctx  = context + (size_t)b * C_ * D_;
    const unsigned rawC_base = lds_off(rawC);

    auto stage_issue = [&](int kbase) {
#pragma unroll
        for (int it = 0; it < 4; ++it) {
            const int e  = it * 256 + t;        // 0..1023
            const int kk = e >> 5;
            const int dd = (e & 31) * 4;
            async_ld16(rawC_base + (unsigned)e * 16u,
                       ctx + (size_t)(kbase + kk) * D_ + db + dd);
        }
    };

    v8f acc[8];
#pragma unroll
    for (int i = 0; i < 8; ++i) acc[i] = (v8f)(0.0f);

    stage_issue(0);                             // pipeline prologue
    for (int k0 = 0; k0 < C_; k0 += 32) {
        const AFrag araw = load_fragA_raw(arow + k0 + hi * 16);  // hoisted global
        wait_async0();
        __syncthreads();                        // old ldsBT fully consumed
        // transpose + pack own chunks into ldsBT[n][k]
#pragma unroll
        for (int it = 0; it < 4; ++it) {
            const int e  = it * 256 + t;
            const int kk = e >> 5;
            const int dd = (e & 31) * 4;
            const float4 v = *(const float4*)(rawC + e * 4);
            ldsBT[(dd + 0) * 40 + kk] = bfu(v.x);
            ldsBT[(dd + 1) * 40 + kk] = bfu(v.y);
            ldsBT[(dd + 2) * 40 + kk] = bfu(v.z);
            ldsBT[(dd + 3) * 40 + kk] = bfu(v.w);
        }
        if (k0 + 32 < C_) stage_issue(k0 + 32); // overlaps with compute below
        __syncthreads();

        const bf16x16 a = pack_fragA(araw);
        bf16x16 bb = load_fragB_lds(ldsBT + lm * 40 + hi * 16);
#pragma unroll
        for (int nt = 0; nt < 8; ++nt) {        // rotate B frag: load nt+1 early
            bf16x16 nxt = bb;
            if (nt < 7)
                nxt = load_fragB_lds(ldsBT + ((nt + 1) * 16 + lm) * 40 + hi * 16);
            acc[nt] = __builtin_amdgcn_wmma_f32_16x16x32_bf16(
                false, a, false, bb, (short)0, acc[nt], false, false);
            bb = nxt;
        }
    }

    // store wcontext: element (m=r+8*hi, n=lm) -> q = w*16+m, d = db+nt*16+n
#pragma unroll
    for (int nt = 0; nt < 8; ++nt) {
#pragma unroll
        for (int r = 0; r < 8; ++r) {
            const int q = w * 16 + r + 8 * hi;
            const int d = db + nt * 16 + lm;
            wctx[((size_t)b * Q_ + q) * D_ + d] = acc[nt][r];
        }
    }
}

extern "C" void kernel_launch(void* const* d_in, const int* in_sizes, int n_in,
                              void* d_out, int out_size, void* d_ws, size_t ws_size,
                              hipStream_t stream) {
    const float* query   = (const float*)d_in[0];   // (B,Q,D)
    const float* context = (const float*)d_in[1];   // (B,C,D)

    float* out_query = (float*)d_out;                          // (B,Q,D)
    float* out_wctx  = out_query + (size_t)B_ * Q_ * D_;       // (B,Q,D)
    float* out_re    = out_wctx  + (size_t)B_ * Q_ * D_;       // (B,Q,C)

    float* attnT = (float*)d_ws;                               // (B,Q,C) f32
    float* norm2 = attnT + (size_t)B_ * Q_ * C_;               // (B,C)   f32

    // output[0] = query (pass-through)
    hipMemcpyAsync(out_query, query, sizeof(float) * (size_t)B_ * Q_ * D_,
                   hipMemcpyDeviceToDevice, stream);

    gemm1_attn<<<dim3(C_ / 128, B_), 256, 0, stream>>>(query, context, attnT, norm2);
    softmax_focal<<<dim3(B_ * Q_), 256, 0, stream>>>(attnT, norm2, out_re);
    gemm2_wctx<<<dim3(D_ / 128, B_), 256, 0, stream>>>(out_re, context, out_wctx);
}
